// WindowMSA_31250182046342
// MI455X (gfx1250) — matrix operand order repaired and checked by reading, not compiled, not run on hardware
//
#include <hip/hip_runtime.h>
#include <hip/hip_fp16.h>

// 3D Swin shifted-window MSA, fully fused. 4 windows per block (weights staged
// once per block), 512 threads = 16 wave32s, async-prefetch of next window's
// x-tile overlapped with current window's attention compute.
// b=2, grid 64^3, E=96, heads=3, dh=32, ws=4 -> 8192 windows -> 2048 blocks.

#define E     96
#define E3    288
#define NTOK  64
#define NZ    64
#define BLOCK 512
#define NWAVE 16
#define WPB   4      // windows per block

#define QKV_STRIDE 200   // halfs: q|k = 192 + 8 pad (400 B rows, 16B aligned)
#define VT_STRIDE  72    // halfs: 64 tokens + 8 pad (144 B rows)
#define XS_STRIDE  104   // halfs: 96 + 8 pad (208 B rows)
#define WQ_STRIDE  104
#define DOT_STRIDE 68    // floats: 64 + 4 pad (272 B rows)
#define ATT_STRIDE 136   // halfs, aliases dots rows in place (272 B rows)
#define OUT_STRIDE 104
#define WP_STRIDE  104

// overlay region (52,224 B): phase1 xs f16 / phase2 dots f32 (attn in place) /
// phase3-4 outS f16
#define OFF_XS    0
#define OFF_DOTS  0
#define OFF_OUTS  0
#define REGION_BYTES 52224

typedef __attribute__((ext_vector_type(16))) _Float16 v16h;
typedef __attribute__((ext_vector_type(8)))  float    v8f;

__device__ __forceinline__ unsigned lds_addr_of(const void* p) {
  return (unsigned)(uintptr_t)(__attribute__((address_space(3))) const char*)p;
}

// ---- WMMA fragment loaders (wave32 layouts, CDNA5 ISA 7.12.2) -------------

// A matrix 16x32 f16: lane = (m = lane&15, half = lane>>4);
// VGPR r<4: K = 2r + 8*half ; r>=4: K = 16 + 2(r-4) + 8*half (pairs packed).
__device__ __forceinline__ v16h load_a_frag(const _Float16* base, int stride,
                                            int row0, int col0, int lane) {
  int m = lane & 15, hh = lane >> 4;
  union { v16h v; unsigned u[8]; } f;
  const _Float16* p = base + (row0 + m) * stride + col0;
#pragma unroll
  for (int r = 0; r < 8; ++r) {
    int k = ((r < 4) ? (2 * r) : (16 + 2 * (r - 4))) + 8 * hh;
    f.u[r] = *(const unsigned*)(p + k);
  }
  return f.v;
}

// B matrix 32x16 f16, storage "n-major": B[k][n] at base[(row0+n)*stride + col0 + k]
// same K order as A: VGPR r<4: K = 2r + 8*half ; r>=4: K = 16 + 2(r-4) + 8*half.
__device__ __forceinline__ v16h load_b_nmajor(const _Float16* base, int stride,
                                              int row0, int col0, int lane) {
  int n = lane & 15, hh = lane >> 4;
  union { v16h v; unsigned u[8]; } f;
  const _Float16* p = base + (row0 + n) * stride + col0;
#pragma unroll
  for (int r = 0; r < 8; ++r) {
    int k = ((r < 4) ? (2 * r) : (16 + 2 * (r - 4))) + 8 * hh;
    f.u[r] = *(const unsigned*)(p + k);
  }
  return f.v;
}

__device__ __forceinline__ v8f wmma_f16(v16h a, v16h b, v8f c) {
  return __builtin_amdgcn_wmma_f32_16x16x32_f16(false, a, false, b,
                                                (short)0, c, false, false);
}

// ---------------------------------------------------------------------------

__global__ __launch_bounds__(BLOCK)
void swin_window_msa_kernel(const float* __restrict__ x,
                            const float* __restrict__ Wqkv,
                            const float* __restrict__ pos,
                            const float* __restrict__ Wproj,
                            const float* __restrict__ bproj,
                            float* __restrict__ out) {
  const int tid  = threadIdx.x;
  const int lane = tid & 31;
  const int wave = tid >> 5;

  // ---- LDS (persistent across the 4-window loop unless noted) ----
  __shared__ __align__(16) _Float16 qkvS[NTOK * QKV_STRIDE];  // per-window q|k
  __shared__ __align__(16) _Float16 vT[E * VT_STRIDE];        // per-window V^T
  __shared__ __align__(16) float    xs32[NTOK * E];           // async landing pad
  __shared__ __align__(16) _Float16 wq[E3 * WQ_STRIDE];       // W_qkv f16
  __shared__ __align__(16) _Float16 wp[E * WP_STRIDE];        // W_proj f16
  __shared__ float posS[343];
  __shared__ float bprojS[96];
  __shared__ __align__(16) unsigned char regionR[REGION_BYTES];

  _Float16* xs   = (_Float16*)(regionR + OFF_XS);
  float*    dots = (float*)(regionR + OFF_DOTS);
  _Float16* attn = (_Float16*)(regionR + OFF_DOTS);  // in-place over dots rows
  _Float16* outS = (_Float16*)(regionR + OFF_OUTS);

  const unsigned xs32_lds = lds_addr_of(xs32);
  const unsigned long long gbase = (unsigned long long)(uintptr_t)x;

  // Issue ASYNCcnt-tracked async copy of window `it_`'s rolled x tile into xs32.
  // roll(x,-2): x_rolled[z'] = x[(z'+2)&63]; output roll(+2) is the same map.
  auto issue_async_x = [&](int it_) {
    int wid_ = blockIdx.x * WPB + it_;
    int wx_ = wid_ & 15, wy_ = (wid_ >> 4) & 15, wz_ = (wid_ >> 8) & 15;
    int bb_ = wid_ >> 12;
    size_t xb = (size_t)bb_ * NZ * NZ * NZ * E;
    for (int c = tid; c < NTOK * 24; c += BLOCK) {   // 64 rows x 24 b128 chunks
      int t = c / 24, q = c - t * 24;
      int tz = t >> 4, ty = (t >> 2) & 3, tx = t & 3;
      int gz = (wz_ * 4 + tz + 2) & 63;
      int gy = (wy_ * 4 + ty + 2) & 63;
      int gx = (wx_ * 4 + tx + 2) & 63;
      unsigned goff =
          (unsigned)((xb + (((size_t)gz * 64 + gy) * 64 + gx) * E) * 4) +
          (unsigned)(q * 16);
      unsigned laddr = xs32_lds + (unsigned)(t * 384 + q * 16);
      asm volatile("global_load_async_to_lds_b128 %0, %1, %2 offset:0"
                   :: "v"(laddr), "v"(goff), "s"(gbase)
                   : "memory");
    }
  };

  // ---- prologue: start window 0's x fetch, then stage all weights once ----
  issue_async_x(0);

  __builtin_prefetch(Wqkv + (size_t)tid * 64, 0, 1);
  for (int i = tid; i < 343; i += BLOCK) posS[i] = pos[i];
  for (int i = tid; i < 96;  i += BLOCK) bprojS[i] = bproj[i];
  for (int i = tid; i < E3 * E; i += BLOCK) {
    int c = i / E, e = i - c * E;
    wq[c * WQ_STRIDE + e] = (_Float16)Wqkv[i];
  }
  for (int i = tid; i < E * E; i += BLOCK) {
    int c = i / E, e = i - c * E;
    wp[c * WP_STRIDE + e] = (_Float16)Wproj[i];
  }

  const float scale = 0.17677669529663687f;  // 32^-0.5

  for (int it = 0; it < WPB; ++it) {
    const int wid = blockIdx.x * WPB + it;
    const int wx = wid & 15, wy = (wid >> 4) & 15, wz = (wid >> 8) & 15;
    const int bb = wid >> 12;
    const size_t xbase = (size_t)bb * NZ * NZ * NZ * E;

    asm volatile("s_wait_asynccnt 0x0" ::: "memory");  // own wave's async done
    __syncthreads();  // all waves' async data visible; prev phase-4 reads done

    // ---- convert staged x tile f32 -> f16 ----
    for (int i = tid; i < NTOK * E; i += BLOCK) {
      int t = i / E, e = i - t * E;
      xs[t * XS_STRIDE + e] = (_Float16)xs32[t * E + e];
    }
    __syncthreads();

    // xs32 consumed: prefetch next window's tile under this window's compute
    if (it + 1 < WPB) issue_async_x(it + 1);

    // ---- qkv = x @ Wqkv^T  (M=64, N=288, K=96) ----
    for (int tIdx = wave; tIdx < 72; tIdx += NWAVE) {
      int Mt = tIdx & 3, Nt = tIdx >> 2;
      v8f c = {};
#pragma unroll
      for (int kk = 0; kk < 3; ++kk) {
        v16h a = load_a_frag(xs, XS_STRIDE, Mt * 16, kk * 32, lane);
        v16h b = load_b_nmajor(wq, WQ_STRIDE, Nt * 16, kk * 32, lane);
        c = wmma_f16(a, b, c);
      }
      int ml = 8 * (lane >> 4), nl = lane & 15;
      if (Nt < 12) {
        // q|k channels -> token-major qkvS
#pragma unroll
        for (int r = 0; r < 8; ++r)
          qkvS[(Mt * 16 + ml + r) * QKV_STRIDE + Nt * 16 + nl] = (_Float16)c[r];
      } else {
        // v channels -> transposed vT[ch][token]; consecutive r = consecutive
        // tokens -> packed b32/b128 stores.
        _Float16* vrow = vT + (Nt * 16 + nl - 192) * VT_STRIDE + Mt * 16 + ml;
#pragma unroll
        for (int r = 0; r < 8; r += 2) {
          union { unsigned u; _Float16 h[2]; } pk;
          pk.h[0] = (_Float16)c[r];
          pk.h[1] = (_Float16)c[r + 1];
          *(unsigned*)(vrow + r) = pk.u;
        }
      }
    }
    __syncthreads();

    // ---- dots = Q K^T * scale + bias + masks  (3 heads, K=dh=32) ----
    for (int tIdx = wave; tIdx < 48; tIdx += NWAVE) {
      int h = tIdx >> 4, rem = tIdx & 15, Mt = rem >> 2, Nt = rem & 3;
      v16h a = load_a_frag(qkvS, QKV_STRIDE, Mt * 16, h * 32, lane);        // Q
      v16h b = load_b_nmajor(qkvS, QKV_STRIDE, Nt * 16, 96 + h * 32, lane); // K^T
      v8f c = {};
      c = wmma_f16(a, b, c);
      int ml = 8 * (lane >> 4), nl = lane & 15;
#pragma unroll
      for (int r = 0; r < 8; ++r) {
        int i = Mt * 16 + ml + r, j = Nt * 16 + nl;
        int iz = i >> 4, iy = (i >> 2) & 3, ix = i & 3;
        int jz = j >> 4, jy = (j >> 2) & 3, jx = j & 3;
        float v = c[r] * scale +
                  posS[(jz - iz + 3) * 49 + (jy - iy + 3) * 7 + (jx - ix + 3)];
        // faithful reference quirk: z- and x-boundary masks both key on x
        bool mz = (wz == 15) && ((ix >= 2) != (jx >= 2));
        bool my = (wy == 15) && ((iy >= 2) != (jy >= 2));
        bool mx = (wx == 15) && ((ix >= 2) != (jx >= 2));
        if (mz | my | mx) v = -__builtin_inff();
        dots[(h * 64 + i) * DOT_STRIDE + j] = v;
      }
    }
    __syncthreads();

    // ---- softmax over j; f16 result written in place over the f32 row ----
    // (per-thread row: write offset 2j always trails read offset 4j -> safe)
    if (tid < 192) {
      int h = tid >> 6, row = tid & 63;
      const float* dr = dots + (h * 64 + row) * DOT_STRIDE;
      float m = -__builtin_inff();
      for (int j = 0; j < 64; ++j) m = fmaxf(m, dr[j]);
      float s = 0.f;
      for (int j = 0; j < 64; ++j) s += __expf(dr[j] - m);
      float inv = 1.f / s;
      _Float16* ar = attn + (h * 64 + row) * ATT_STRIDE;
      for (int j = 0; j < 64; ++j) {
        float e = __expf(dr[j] - m) * inv;
        ar[j] = (_Float16)e;
      }
    }
    __syncthreads();

    // ---- out = attn @ V  (per head: M=64, N=32, K=64) ----
    for (int tIdx = wave; tIdx < 24; tIdx += NWAVE) {
      int h = tIdx >> 3, rem = tIdx & 7, Mt = rem >> 1, Nt = rem & 1;
      v8f c = {};
#pragma unroll
      for (int kk = 0; kk < 2; ++kk) {
        v16h a = load_a_frag(attn, ATT_STRIDE, h * 64 + Mt * 16, kk * 32, lane);
        v16h b = load_b_nmajor(vT, VT_STRIDE, h * 32 + Nt * 16, kk * 32, lane);
        c = wmma_f16(a, b, c);
      }
      int ml = 8 * (lane >> 4), nl = lane & 15;
#pragma unroll
      for (int r = 0; r < 8; ++r)
        outS[(Mt * 16 + ml + r) * OUT_STRIDE + h * 32 + Nt * 16 + nl] =
            (_Float16)c[r];
    }
    __syncthreads();

    // ---- final = out @ Wproj^T + b_proj, write un-rolled f32 ----
    for (int tIdx = wave; tIdx < 24; tIdx += NWAVE) {
      int Mt = tIdx & 3, Nt = tIdx >> 2;  // Nt in 0..5
      v8f c = {};
#pragma unroll
      for (int kk = 0; kk < 3; ++kk) {
        v16h a = load_a_frag(outS, OUT_STRIDE, Mt * 16, kk * 32, lane);
        v16h b = load_b_nmajor(wp, WP_STRIDE, Nt * 16, kk * 32, lane);
        c = wmma_f16(a, b, c);
      }
      int ml = 8 * (lane >> 4), nl = lane & 15;
#pragma unroll
      for (int r = 0; r < 8; ++r) {
        int t = Mt * 16 + ml + r;
        int cch = Nt * 16 + nl;
        int tz = t >> 4, ty = (t >> 2) & 3, tx = t & 3;
        int gz = (wz * 4 + tz + 2) & 63;
        int gy = (wy * 4 + ty + 2) & 63;
        int gx = (wx * 4 + tx + 2) & 63;
        out[xbase + (((size_t)gz * 64 + gy) * 64 + gx) * E + cch] =
            c[r] + bprojS[cch];
      }
    }
    // next iteration's s_wait_asynccnt + __syncthreads orders outS/region reuse
  }
}

extern "C" void kernel_launch(void* const* d_in, const int* in_sizes, int n_in,
                              void* d_out, int out_size, void* d_ws, size_t ws_size,
                              hipStream_t stream) {
  (void)in_sizes; (void)n_in; (void)out_size; (void)d_ws; (void)ws_size;
  const float* x     = (const float*)d_in[0];
  const float* Wqkv  = (const float*)d_in[1];
  const float* pos   = (const float*)d_in[2];
  const float* Wproj = (const float*)d_in[3];
  const float* bproj = (const float*)d_in[4];
  float* out = (float*)d_out;

  dim3 grid(2 * 16 * 16 * 16 / WPB);  // 4 windows per block
  dim3 block(BLOCK);                  // 16 wave32s
  swin_window_msa_kernel<<<grid, block, 0, stream>>>(x, Wqkv, pos, Wproj,
                                                     bproj, out);
}
